// MultiHeadAttention_48120813585129
// MI455X (gfx1250) — matrix operand order repaired and checked by reading, not compiled, not run on hardware
//
#include <hip/hip_runtime.h>
#include <hip/hip_bf16.h>

// ---------------------------------------------------------------------------
// Causal linear attention (chunked) for MI455X / gfx1250, wave32 + WMMA bf16.
//   N=2, L=2048, DIM=1024, H=16, Dh=64, chunk C=64, NC=32, NH=N*H=32
// ---------------------------------------------------------------------------

#define DIMX  1024
#define HEADS 16
#define DH    64
#define LSEQ  2048
#define NBAT  2
#define CHUNK 64
#define NCH   (LSEQ / CHUNK)   // 32
#define NHEAD (NBAT * HEADS)   // 32
#define MROWS (NBAT * LSEQ)    // 4096

typedef __attribute__((ext_vector_type(16))) __bf16 bf16x16;
typedef __attribute__((ext_vector_type(8)))  __bf16 bf16x8;
typedef __attribute__((ext_vector_type(8)))  float  f32x8;

// ---- fragment loaders --------------------------------------------------
// A-frag (16x32 MxK, 16-bit): lane holds row m=lane&15.
//   half=0: elems 0-7 -> K k0+0..7,   elems 8-15 -> K k0+16..23
//   half=1: elems 0-7 -> K k0+8..15,  elems 8-15 -> K k0+24..31
__device__ __forceinline__ bf16x16 load_a_frag(const __bf16* row_ptr, int k0, int half) {
  const __bf16* p = row_ptr + k0 + half * 8;
  bf16x8 lo = *(const bf16x8*)(p);
  bf16x8 hi = *(const bf16x8*)(p + 16);
  bf16x16 f;
#pragma unroll
  for (int i = 0; i < 8; ++i) { f[i] = lo[i]; f[i + 8] = hi[i]; }
  return f;
}

// B-frag (32x16 KxN, 16-bit) loaded from an N x K row-major source
// (i.e. the "transposed" operand): lane holds col n=lane&15.
//   half=0: elems 0-7 -> K k0+0..7,   elems 8-15 -> K k0+16..23
//   half=1: elems 0-7 -> K k0+8..15,  elems 8-15 -> K k0+24..31
__device__ __forceinline__ bf16x16 load_b_frag(const __bf16* row_ptr, int k0, int half) {
  const __bf16* p = row_ptr + k0 + half * 8;
  bf16x8 lo = *(const bf16x8*)(p);
  bf16x8 hi = *(const bf16x8*)(p + 16);
  bf16x16 f;
#pragma unroll
  for (int i = 0; i < 8; ++i) { f[i] = lo[i]; f[i + 8] = hi[i]; }
  return f;
}

__device__ __forceinline__ f32x8 wmma_bf16(bf16x16 a, bf16x16 b, f32x8 c) {
  return __builtin_amdgcn_wmma_f32_16x16x32_bf16(false, a, false, b, (short)0, c,
                                                 false, false);
}

// ---- kernel 0: f32 -> bf16 convert -------------------------------------
__global__ __launch_bounds__(256) void k0_cvt(const float* __restrict__ in,
                                              __bf16* __restrict__ out, int n) {
  int i = blockIdx.x * 256 + threadIdx.x;
  if (i < n) out[i] = (__bf16)in[i];
}

// ---- kernel 1: projections + fused per-head softmax --------------------
// grid = 3 mats * 16 mblocks * 16 heads = 768 blocks, 256 thr (8 waves).
// Block computes 256 rows x 64 cols (one head).  The head's weight k-chunk
// (64x64 bf16 = 8KB) is staged in LDS, double buffered, one barrier per
// chunk, with next-chunk global loads issued before the WMMA burst.
// Each wave owns a 32x64 tile: 8 f32 accumulators, 8 WMMA per k-step,
// with all 4 LDS B-frags hoisted ahead of the WMMA burst.
__global__ __launch_bounds__(256) void k1_proj(
    const __bf16* __restrict__ xq, const __bf16* __restrict__ xk,
    const __bf16* __restrict__ wq, const __bf16* __restrict__ wk,
    const __bf16* __restrict__ wv, __bf16* __restrict__ qs,
    __bf16* __restrict__ ks, __bf16* __restrict__ ksT,
    __bf16* __restrict__ vT) {
  __shared__ __bf16 Bsm[2][64 * 64];   // 2 x 8KB

  int bid  = blockIdx.x;
  int mat  = bid >> 8;        // /256 : 0:q 1:k 2:v
  int rem  = bid & 255;
  int mblk = rem >> 4;        // 0..15 (256-row block)
  int h    = rem & 15;
  int tid  = threadIdx.x;
  int wave = tid >> 5;
  int lane = tid & 31;
  int half = lane >> 4;
  int l16  = lane & 15;

  const __bf16* X = (mat == 0) ? xq : xk;
  const __bf16* W = (mat == 0) ? wq : ((mat == 1) ? wk : wv);

  int rowbase = mblk * 256 + wave * 32;  // this wave's 32 output rows
  const __bf16* arow0 = X + (size_t)(rowbase + l16) * DIMX;
  const __bf16* arow1 = X + (size_t)(rowbase + 16 + l16) * DIMX;

  // LDS fill: 64 rows x 64 cols bf16 per chunk = 512 x 16B units, 2/thread.
  int u0 = tid, u1 = tid + 256;
  const __bf16* wsrc0 = W + (size_t)(h * 64 + (u0 >> 3)) * DIMX + (u0 & 7) * 8;
  const __bf16* wsrc1 = W + (size_t)(h * 64 + (u1 >> 3)) * DIMX + (u1 & 7) * 8;
  __bf16* wdst0 = &Bsm[0][(u0 >> 3) * 64 + (u0 & 7) * 8];
  __bf16* wdst1 = &Bsm[0][(u1 >> 3) * 64 + (u1 & 7) * 8];

  // prologue: fill buffer 0 with chunk 0
  bf16x8 pre0 = *(const bf16x8*)(wsrc0);
  bf16x8 pre1 = *(const bf16x8*)(wsrc1);
  *(bf16x8*)(wdst0) = pre0;
  *(bf16x8*)(wdst1) = pre1;
  __syncthreads();

  f32x8 acc[2][4] = {};
  for (int kc = 0; kc < 16; ++kc) {
    int buf = kc & 1;
    bool more = (kc + 1) < 16;
    if (more) {  // issue next chunk's global loads early (overlap with WMMA)
      pre0 = *(const bf16x8*)(wsrc0 + (kc + 1) * 64);
      pre1 = *(const bf16x8*)(wsrc1 + (kc + 1) * 64);
    }
#pragma unroll
    for (int ksi = 0; ksi < 2; ++ksi) {
      int ks = ksi * 32;
      bf16x16 a0 = load_a_frag(arow0, kc * 64 + ks, half);
      bf16x16 a1 = load_a_frag(arow1, kc * 64 + ks, half);
      bf16x16 bfr[4];
#pragma unroll
      for (int t = 0; t < 4; ++t)
        bfr[t] = load_b_frag(&Bsm[buf][(t * 16 + l16) * 64], ks, half);
#pragma unroll
      for (int t = 0; t < 4; ++t) {
        acc[0][t] = wmma_bf16(a0, bfr[t], acc[0][t]);
        acc[1][t] = wmma_bf16(a1, bfr[t], acc[1][t]);
      }
    }
    if (more) {  // store into the *other* buffer: no race with this chunk's
                 // readers; previous readers of buf^1 are behind last barrier
      *(bf16x8*)(wdst0 + (buf ^ 1) * 64 * 64) = pre0;
      *(bf16x8*)(wdst1 + (buf ^ 1) * 64 * 64) = pre1;
    }
    __syncthreads();
  }

  if (mat < 2) {
    // softmax over the 64 head columns, row-wise, independently for both
    // 16-row groups.  Row m = r + 8*half lives in this half's 16 lanes.
#pragma unroll
    for (int g = 0; g < 2; ++g) {
#pragma unroll
      for (int r = 0; r < 8; ++r) {
        float mx = acc[g][0][r];
#pragma unroll
        for (int t = 1; t < 4; ++t) mx = fmaxf(mx, acc[g][t][r]);
#pragma unroll
        for (int s = 1; s < 16; s <<= 1) mx = fmaxf(mx, __shfl_xor(mx, s, 32));
        float sum = 0.f;
#pragma unroll
        for (int t = 0; t < 4; ++t) {
          float e = __expf(acc[g][t][r] - mx);
          acc[g][t][r] = e;
          sum += e;
        }
#pragma unroll
        for (int s = 1; s < 16; s <<= 1) sum += __shfl_xor(sum, s, 32);
        float inv = 1.0f / sum;
#pragma unroll
        for (int t = 0; t < 4; ++t) acc[g][t][r] *= inv;
      }
    }
  }

#pragma unroll
  for (int g = 0; g < 2; ++g)
#pragma unroll
    for (int r = 0; r < 8; ++r) {
      int gm = rowbase + g * 16 + r + 8 * half;
      int nb = gm >> 11;  // /2048
      int l  = gm & 2047;
      int nh = nb * 16 + h;
#pragma unroll
      for (int t = 0; t < 4; ++t) {
        int d = t * 16 + l16;
        __bf16 v = (__bf16)acc[g][t][r];
        if (mat == 0) {
          qs[((size_t)nh * LSEQ + l) * DH + d] = v;
        } else if (mat == 1) {
          ks[((size_t)nh * LSEQ + l) * DH + d] = v;
          ksT[((size_t)nh * DH + d) * LSEQ + l] = v;
        } else {
          vT[((size_t)nh * DH + d) * LSEQ + l] = v;
        }
      }
    }
}

// ---- kernel 2: per-chunk state G[c] = K_c^T V_c ------------------------
// 1024 chunk-heads, 2 per block; 4 waves per chunk-head (one 16x64 row band).
__global__ __launch_bounds__(256) void k2_state(const __bf16* __restrict__ ksT,
                                                const __bf16* __restrict__ vT,
                                                float* __restrict__ G) {
  int wave = threadIdx.x >> 5, lane = threadIdx.x & 31;
  int half = lane >> 4, l16 = lane & 15;
  int idx  = blockIdx.x * 2 + (wave >> 2);  // chunk-head id
  int mrow = wave & 3;
  int nh = idx >> 5;
  int c  = idx & 31;

  const __bf16* abase = ksT + (size_t)nh * DH * LSEQ;  // A[i][t] = K^T, row stride L
  const __bf16* bbase = vT + (size_t)nh * DH * LSEQ;   // B from V^T (NxK row-major)
  const __bf16* arow = abase + (size_t)(mrow * 16 + l16) * LSEQ + c * 64;

  f32x8 acc[4] = {};
#pragma unroll
  for (int k0 = 0; k0 < 64; k0 += 32) {
    bf16x16 a = load_a_frag(arow, k0, half);
    bf16x16 bfr[4];
#pragma unroll
    for (int t = 0; t < 4; ++t)
      bfr[t] = load_b_frag(bbase + (size_t)(t * 16 + l16) * LSEQ + c * 64, k0, half);
#pragma unroll
    for (int t = 0; t < 4; ++t) acc[t] = wmma_bf16(a, bfr[t], acc[t]);
  }
  float* g = G + (size_t)idx * 4096;
#pragma unroll
  for (int t = 0; t < 4; ++t)
#pragma unroll
    for (int r = 0; r < 8; ++r) {
      int i = mrow * 16 + r + 8 * half;
      int j = t * 16 + l16;
      g[i * 64 + j] = acc[t][r];
    }
}

// ---- kernel 3: exclusive prefix scan of G over chunks, store S^T bf16 --
__global__ __launch_bounds__(256) void k3_scan(const float* __restrict__ G,
                                               __bf16* __restrict__ ST) {
  int nh = blockIdx.x;
  int tid = threadIdx.x;
  float run[16];
#pragma unroll
  for (int u = 0; u < 16; ++u) run[u] = 0.f;
  for (int c = 0; c < NCH; ++c) {
    const float* g = G + ((size_t)nh * NCH + c) * 4096;
    __bf16* st = ST + ((size_t)nh * NCH + c) * 4096;
#pragma unroll
    for (int u = 0; u < 16; ++u) {
      int e = tid + u * 256;
      int i = e >> 6, j = e & 63;
      st[j * 64 + i] = (__bf16)run[u];  // store transposed (row j, col i)
      run[u] += g[e];
    }
  }
}

// ---- kernel 4: O = mask(QK^T) V + Q S_prev -----------------------------
__global__ __launch_bounds__(256) void k4_out(const __bf16* __restrict__ qs,
                                              const __bf16* __restrict__ ks,
                                              const __bf16* __restrict__ vT,
                                              const __bf16* __restrict__ ST,
                                              float* __restrict__ out) {
  __shared__ __bf16 Pls[8][16 * 64];  // per-wave P row band, 16KB total
  int wave = threadIdx.x >> 5, lane = threadIdx.x & 31;
  int half = lane >> 4, l16 = lane & 15;
  int idx  = blockIdx.x * 2 + (wave >> 2);
  int mrow = wave & 3;
  int nh = idx >> 5, c = idx & 31;
  int nb = nh >> 4, h = nh & 15;

  const __bf16* qbase = qs + ((size_t)nh * LSEQ + (size_t)c * 64) * DH;
  const __bf16* kbase = ks + ((size_t)nh * LSEQ + (size_t)c * 64) * DH;
  const __bf16* vtb = vT + (size_t)nh * DH * LSEQ;
  const __bf16* stb = ST + ((size_t)nh * NCH + c) * 4096;
  const __bf16* qrow = qbase + (size_t)(mrow * 16 + l16) * DH;

  // P = Q K^T  (reduce over d)
  f32x8 pacc[4] = {};
#pragma unroll
  for (int k0 = 0; k0 < 64; k0 += 32) {
    bf16x16 a = load_a_frag(qrow, k0, half);
    bf16x16 bfr[4];
#pragma unroll
    for (int sn = 0; sn < 4; ++sn)
      bfr[sn] = load_b_frag(kbase + (size_t)(sn * 16 + l16) * DH, k0, half);
#pragma unroll
    for (int sn = 0; sn < 4; ++sn) pacc[sn] = wmma_bf16(a, bfr[sn], pacc[sn]);
  }
  // causal mask within chunk, convert to bf16, stage C-layout -> A-layout in LDS
#pragma unroll
  for (int sn = 0; sn < 4; ++sn)
#pragma unroll
    for (int r = 0; r < 8; ++r) {
      int t = mrow * 16 + r + 8 * half;
      int s = sn * 16 + l16;
      float v = (s <= t) ? pacc[sn][r] : 0.f;
      Pls[wave][(r + 8 * half) * 64 + s] = (__bf16)v;
    }
  __syncthreads();

  // O = P V + Q S_prev  (both reduce over 64 in two k-steps)
  f32x8 oacc[4] = {};
#pragma unroll
  for (int k0 = 0; k0 < 64; k0 += 32) {
    bf16x16 ap = load_a_frag(&Pls[wave][l16 * 64], k0, half);
    bf16x16 aq = load_a_frag(qrow, k0, half);
    bf16x16 bv[4], bs[4];
#pragma unroll
    for (int jn = 0; jn < 4; ++jn) {
      bv[jn] = load_b_frag(vtb + (size_t)(jn * 16 + l16) * LSEQ + c * 64, k0, half);
      bs[jn] = load_b_frag(stb + (size_t)(jn * 16 + l16) * 64, k0, half);
    }
#pragma unroll
    for (int jn = 0; jn < 4; ++jn) {
      oacc[jn] = wmma_bf16(ap, bv[jn], oacc[jn]);
      oacc[jn] = wmma_bf16(aq, bs[jn], oacc[jn]);
    }
  }
#pragma unroll
  for (int jn = 0; jn < 4; ++jn)
#pragma unroll
    for (int r = 0; r < 8; ++r) {
      int t = mrow * 16 + r + 8 * half;
      int l = c * 64 + t;
      int d = jn * 16 + l16;
      out[((size_t)nb * LSEQ + l) * DIMX + h * 64 + d] = oacc[jn][r];
    }
}

// ---- host launch -------------------------------------------------------
extern "C" void kernel_launch(void* const* d_in, const int* in_sizes, int n_in,
                              void* d_out, int out_size, void* d_ws, size_t ws_size,
                              hipStream_t stream) {
  (void)in_sizes; (void)n_in; (void)out_size; (void)ws_size;
  const float* query = (const float*)d_in[0];
  const float* key   = (const float*)d_in[1];
  const float* Wq    = (const float*)d_in[2];
  const float* Wk    = (const float*)d_in[3];
  const float* Wv    = (const float*)d_in[4];
  float* out = (float*)d_out;

  char* ws = (char*)d_ws;
  size_t off = 0;
  auto carve = [&](size_t bytes) -> void* {
    void* p = ws + off;
    off += (bytes + 255) & ~(size_t)255;
    return p;
  };
  const size_t act_elems = (size_t)NHEAD * LSEQ * DH;  // 4.19M
  __bf16* xqb = (__bf16*)carve((size_t)MROWS * DIMX * 2);
  __bf16* xkb = (__bf16*)carve((size_t)MROWS * DIMX * 2);
  __bf16* wqb = (__bf16*)carve((size_t)DIMX * DIMX * 2);
  __bf16* wkb = (__bf16*)carve((size_t)DIMX * DIMX * 2);
  __bf16* wvb = (__bf16*)carve((size_t)DIMX * DIMX * 2);
  __bf16* qsb = (__bf16*)carve(act_elems * 2);
  __bf16* ksb = (__bf16*)carve(act_elems * 2);
  __bf16* ksT = (__bf16*)carve(act_elems * 2);
  __bf16* vTb = (__bf16*)carve(act_elems * 2);
  float*  Gbuf = (float*)carve((size_t)NHEAD * NCH * 4096 * 4);
  __bf16* STb = (__bf16*)carve((size_t)NHEAD * NCH * 4096 * 2);

  int nx = MROWS * DIMX;  // 4,194,304
  int nw = DIMX * DIMX;   // 1,048,576
  k0_cvt<<<(nx + 255) / 256, 256, 0, stream>>>(query, xqb, nx);
  k0_cvt<<<(nx + 255) / 256, 256, 0, stream>>>(key, xkb, nx);
  k0_cvt<<<(nw + 255) / 256, 256, 0, stream>>>(Wq, wqb, nw);
  k0_cvt<<<(nw + 255) / 256, 256, 0, stream>>>(Wk, wkb, nw);
  k0_cvt<<<(nw + 255) / 256, 256, 0, stream>>>(Wv, wvb, nw);

  k1_proj<<<768, 256, 0, stream>>>(xqb, xkb, wqb, wkb, wvb, qsb, ksb, ksT, vTb);
  k2_state<<<512, 256, 0, stream>>>(ksT, vTb, Gbuf);
  k3_scan<<<NHEAD, 256, 0, stream>>>(Gbuf, STb);
  k4_out<<<512, 256, 0, stream>>>(qsb, ksb, vTb, STb, out);
}
